// EfficientVectorQuantizer_17721035063477
// MI455X (gfx1250) — matrix operand order repaired and checked by reading, not compiled, not run on hardware
//
#include <hip/hip_runtime.h>

typedef __attribute__((ext_vector_type(16))) _Float16 v16h;
typedef __attribute__((ext_vector_type(8)))  float    v8f;
typedef __attribute__((ext_vector_type(4)))  float    v4f;

#define N_EMB       1024
#define CDIM        64
#define HW          4096          // h*w
#define BC_STRIDE   262144        // c*h*w
#define NTOT        131072        // b*h*w
#define OUT_ELEMS   8388608       // b*c*h*w
#define ROWS_PER_BLOCK 256
#define NBLOCKS     (NTOT / ROWS_PER_BLOCK)   // 512

// ---------------- Kernel 0: code norms + f16 codebook ----------------
__global__ void vq_prep(const float* __restrict__ emb,
                        _Float16* __restrict__ cb16,
                        float* __restrict__ enorm) {
  int k = blockIdx.x * blockDim.x + threadIdx.x;
  if (k < N_EMB) {
    const float* row = emb + k * CDIM;
    _Float16* drow = cb16 + k * CDIM;
    float s = 0.f;
#pragma unroll
    for (int c = 0; c < CDIM; ++c) {
      float v = row[c];
      s = fmaf(v, v, s);
      drow[c] = (_Float16)v;
    }
    enorm[k] = s;
  }
}

// ---------------- Kernel 1: WMMA distance + argmin + scatter ----------------
__launch_bounds__(256)
__global__ void vq_main(const float* __restrict__ x,
                        const float* __restrict__ emb,
                        const _Float16* __restrict__ cb16,
                        const float* __restrict__ enorm_g,
                        float* __restrict__ out,
                        float* __restrict__ partials) {
  extern __shared__ char smem[];
  _Float16* cb    = (_Float16*)smem;                                   // 128 KB
  float*    enorm = (float*)(smem + N_EMB * CDIM * 2);                 //   4 KB
  int*      idxs  = (int*)(smem + N_EMB * CDIM * 2 + N_EMB * 4);       //   1 KB
  float*    wred  = (float*)(smem + N_EMB * CDIM * 2 + N_EMB * 4 + ROWS_PER_BLOCK * 4);

  const int tid   = threadIdx.x;
  const int wave  = tid >> 5;
  const int lane  = tid & 31;
  const int lhalf = lane >> 4;        // which 16-lane half
  const int lm    = lane & 15;

  const int n0   = blockIdx.x * ROWS_PER_BLOCK;   // 4096 % 256 == 0: block stays in one image
  const int bimg = n0 >> 12;
  const int rem  = n0 & 4095;
  const float* xbase = x + (size_t)bimg * BC_STRIDE + rem;   // + ch*4096 + row_in_block

  // A operands (contiguous k = 16*half + s, same order as B) + f32 x-norm partials for the loss.
  // Lanes l and l+16 hold complementary channel halves of row lm, so summing the
  // plain per-lane partials over all 32 lanes yields the wave's total sum ||x||^2.
  v16h A[2][2];
  float xnp = 0.f;
#pragma unroll
  for (int a = 0; a < 2; ++a) {
    const int rr = wave * 32 + a * 16 + lm;       // row-in-block, M = lm
#pragma unroll
    for (int ks = 0; ks < 2; ++ks) {
      const int ch0 = ks * 32 + lhalf * 16;       // K groups: {0-15} or {16-31}
      v16h av;
#pragma unroll
      for (int e = 0; e < 8; ++e) {
        float v0 = xbase[(ch0 + e) * HW + rr];
        float v1 = xbase[(ch0 + 8 + e) * HW + rr];
        xnp = fmaf(v0, v0, xnp);
        xnp = fmaf(v1, v1, xnp);
        av[e]     = (_Float16)v0;
        av[e + 8] = (_Float16)v1;
      }
      A[a][ks] = av;
    }
  }

  // Stage full codebook (f16) + norms into LDS (overlaps with x loads above).
  {
    const v4f* src = (const v4f*)cb16;           // 8192 x 16B
    v4f* dst = (v4f*)cb;
#pragma unroll 4
    for (int i = tid; i < (N_EMB * CDIM * 2) / 16; i += 256) dst[i] = src[i];
    ((v4f*)enorm)[tid] = ((const v4f*)enorm_g)[tid];   // 1024 floats = 256 v4f
  }
  __syncthreads();

  float minv[2][8];
  int   mini[2][8];
#pragma unroll
  for (int a = 0; a < 2; ++a)
#pragma unroll
    for (int v = 0; v < 8; ++v) { minv[a][v] = 3.4e38f; mini[a][v] = 0; }

  // Sweep all 64 code tiles; K=64 via two chained WMMAs per accumulator.
  for (int t = 0; t < N_EMB / 16; ++t) {
    const int code = t * 16 + lm;                 // N = lane%16 in C layout
    const float en = enorm[code];
    const _Float16* brow = cb + code * CDIM;      // B: lanes0-15 K=0..15, lanes16-31 K=16..31
    v16h B0 = *(const v16h*)(brow + lhalf * 16);        // K-tile 0 (ch 0..31)
    v16h B1 = *(const v16h*)(brow + 32 + lhalf * 16);   // K-tile 1 (ch 32..63)
#pragma unroll
    for (int a = 0; a < 2; ++a) {
      v8f c = {};
      c = __builtin_amdgcn_wmma_f32_16x16x32_f16(false, A[a][0], false, B0, (short)0, c, false, false);
      c = __builtin_amdgcn_wmma_f32_16x16x32_f16(false, A[a][1], false, B1, (short)0, c, false, false);
#pragma unroll
      for (int v = 0; v < 8; ++v) {
        float d = fmaf(-2.f, c[v], en);           // ||e||^2 - 2 x.e  (||x||^2 constant per row)
        bool p = d < minv[a][v];
        minv[a][v] = p ? d : minv[a][v];
        mini[a][v] = p ? code : mini[a][v];
      }
    }
  }

  // Per-row argmin: butterfly across the 16 lanes of each half (xor<16 preserves bit4).
  // Loss accumulation: sum(x-e)^2 = sum ||x||^2 + sum min(||e||^2 - 2 x.e).
  float wloss = xnp;
#pragma unroll
  for (int a = 0; a < 2; ++a) {
#pragma unroll
    for (int v = 0; v < 8; ++v) {
      float mv = minv[a][v];
      int   mi = mini[a][v];
#pragma unroll
      for (int off = 1; off < 16; off <<= 1) {
        float ov = __shfl_xor(mv, off, 32);
        int   oi = __shfl_xor(mi, off, 32);
        bool p = (ov < mv) || (ov == mv && oi < mi);   // first-index tie-break
        mv = p ? ov : mv;
        mi = p ? oi : mi;
      }
      minv[a][v] = mv;
      mini[a][v] = mi;
    }
    if (lm == 0) {                                // lane 0: rows v, lane 16: rows v+8
#pragma unroll
      for (int v = 0; v < 8; ++v) {
        idxs[wave * 32 + a * 16 + lhalf * 8 + v] = mini[a][v];
        wloss += minv[a][v];                      // per-row min, 8 rows per contributing lane
      }
    }
  }
  // Wave-level loss reduction (lanes 0/16 carry the min sums, all lanes carry xnorm).
#pragma unroll
  for (int off = 16; off >= 1; off >>= 1) wloss += __shfl_xor(wloss, off, 32);
  if (lane == 0) wred[wave] = wloss;
  __syncthreads();
  if (tid == 0) {
    float s = 0.f;
#pragma unroll
    for (int w = 0; w < 8; ++w) s += wred[w];
    partials[blockIdx.x] = s;
  }

  // Phase 2: gather fp32 code rows and scatter to NCHW layout (coalesced per channel).
  const int k = idxs[tid];
  const float* erow = emb + k * CDIM;
  float* obase = out + (size_t)bimg * BC_STRIDE + rem;
#pragma unroll 8
  for (int c = 0; c < CDIM; ++c) {
    obase[c * HW + tid] = erow[c];
  }
}

// ---------------- Kernel 2: deterministic loss reduction ----------------
__global__ void vq_loss(const float* __restrict__ partials, float* __restrict__ loss) {
  __shared__ float wred[8];
  const int tid = threadIdx.x;
  float s = partials[tid] + partials[tid + 256];
#pragma unroll
  for (int off = 16; off >= 1; off >>= 1) s += __shfl_xor(s, off, 32);
  if ((tid & 31) == 0) wred[tid >> 5] = s;
  __syncthreads();
  if (tid == 0) {
    float t = 0.f;
#pragma unroll
    for (int w = 0; w < 8; ++w) t += wred[w];
    *loss = t * (1.25f / (float)OUT_ELEMS);       // mean + 0.25*mean of same MSE
  }
}

extern "C" void kernel_launch(void* const* d_in, const int* in_sizes, int n_in,
                              void* d_out, int out_size, void* d_ws, size_t ws_size,
                              hipStream_t stream) {
  (void)in_sizes; (void)n_in; (void)ws_size;
  const float* x   = (const float*)d_in[0];
  const float* emb = (const float*)d_in[1];
  float* out  = (float*)d_out;
  float* loss = out + (out_size - 1);

  _Float16* cb16  = (_Float16*)d_ws;                                  // 128 KB
  float* enorm    = (float*)((char*)d_ws + N_EMB * CDIM * 2);         //   4 KB
  float* partials = (float*)((char*)d_ws + N_EMB * CDIM * 2 + N_EMB * 4); // 2 KB

  size_t shmem = N_EMB * CDIM * 2 + N_EMB * 4 + ROWS_PER_BLOCK * 4 + 8 * 4;  // ~133 KB (< 320 KB/WGP)
  (void)hipFuncSetAttribute((const void*)vq_main,
                            hipFuncAttributeMaxDynamicSharedMemorySize, (int)shmem);

  vq_prep<<<N_EMB / 256, 256, 0, stream>>>(emb, cb16, enorm);
  vq_main<<<NBLOCKS, 256, shmem, stream>>>(x, emb, cb16, enorm, out, partials);
  vq_loss<<<1, 256, 0, stream>>>(partials, loss);
}
